// GraphReasoningModule_79164837200290
// MI455X (gfx1250) — compile-verified, matrix operand order checked by reading, not run on hardware
//
#include <hip/hip_runtime.h>
#include <hip/hip_bf16.h>
#include <math.h>

typedef __attribute__((ext_vector_type(16))) _Float16 v16h;
typedef __attribute__((ext_vector_type(8)))  _Float16 v8h;
typedef __attribute__((ext_vector_type(8)))  float    v8f;

#define B_SZ 8
#define L_SZ 8192
#define D_SZ 128
#define E_SZ 64
#define NB_STRIDE 910   /* max(1, L // (k_stride+1)) = 8192//9 */

// ---- LDS layout (bytes), all 16B aligned ----
#define OFF_WE1S 0        /* 64x128 f16  16384 */
#define OFF_WE1N 16384    /* 64x128 f16  16384 */
#define OFF_WE2  32768    /* 64x64  f16   8192 */
#define OFF_WMN  40960    /* 128x128 f16 32768 */
#define OFF_WME  73728    /* 128x64 f16  16384 */
#define OFF_WU   0        /* 256x256 f16 131072 (phase-2 overlay) */
#define OFF_HN   131072   /* 256x128 f16 65536 */
#define OFF_HS   196608   /* 16x128 f16   4096 */
#define OFF_ES   200704   /* 16x64  f16   2048 */
#define OFF_EG   202752   /* 256x64 f16  32768 */
#define OFF_ED   235520   /* 256x64 f16  32768 ; reused as 16x256 f32 upd */
#define OFF_AT   268288   /* 16x16  f32   1024 */
#define OFF_CA   269312   /* 16x256 f16   8192 */
#define OFF_BIAS 277504   /* 2560 */
#define SMEM_TOTAL 280064

__device__ __forceinline__ float gelu_exact(float x) {
    return 0.5f * x * (1.0f + erff(x * 0.70710678118654752440f));
}

__device__ __forceinline__ int nb_index(int l, int j) {
    if (j < 8) {
        int off = (j < 4) ? (j - 4) : (j - 3);   // -4,-3,-2,-1,1,2,3,4
        int t = l + off;
        return t < 0 ? 0 : (t > L_SZ - 1 ? L_SZ - 1 : t);
    }
    return (j - 8) * NB_STRIDE;                  // strided anchors
}

// Load a 16x32 f16 fragment (A or B^T) from an LDS tile stored row-major,
// matching the CDNA5 16-bit A/B VGPR layout:
// row = lane&15 ; K groups: (lane>>4)*8 .. +7 and 16+(lane>>4)*8 .. +7
__device__ __forceinline__ v16h ld_frag(const _Float16* base, int stride, int kc, int lane) {
    const _Float16* p = base + (lane & 15) * stride + kc * 32 + ((lane >> 4) << 3);
    v8h lo = *(const v8h*)(p);
    v8h hi = *(const v8h*)(p + 16);
    v16h r;
#pragma unroll
    for (int i = 0; i < 8; ++i) { r[i] = lo[i]; r[8 + i] = hi[i]; }
    return r;
}

__device__ __forceinline__ v8f wmma_step(v16h a, v16h b, v8f c) {
    return __builtin_amdgcn_wmma_f32_16x16x32_f16(false, a, false, b, (short)0, c, false, false);
}

__global__ __launch_bounds__(256)
void graph_iter_kernel(const float* __restrict__ hin, float* __restrict__ hout,
                       const float* __restrict__ w_e1, const float* __restrict__ b_e1,
                       const float* __restrict__ w_e2, const float* __restrict__ b_e2,
                       const float* __restrict__ w_m,  const float* __restrict__ b_m,
                       const float* __restrict__ w_a,  const float* __restrict__ b_a,
                       const float* __restrict__ w_u,  const float* __restrict__ b_u,
                       const float* __restrict__ g_un, const float* __restrict__ b_un) {
    extern __shared__ char smem[];
    _Float16* sWe1s = (_Float16*)(smem + OFF_WE1S);
    _Float16* sWe1n = (_Float16*)(smem + OFF_WE1N);
    _Float16* sWe2  = (_Float16*)(smem + OFF_WE2);
    _Float16* sWmn  = (_Float16*)(smem + OFF_WMN);
    _Float16* sWme  = (_Float16*)(smem + OFF_WME);
    _Float16* sWu   = (_Float16*)(smem + OFF_WU);
    _Float16* sHN   = (_Float16*)(smem + OFF_HN);
    _Float16* sHS   = (_Float16*)(smem + OFF_HS);
    _Float16* sES   = (_Float16*)(smem + OFF_ES);
    _Float16* sEG   = (_Float16*)(smem + OFF_EG);
    _Float16* sED   = (_Float16*)(smem + OFF_ED);
    float*    sAT   = (float*)   (smem + OFF_AT);
    _Float16* sCA   = (_Float16*)(smem + OFF_CA);
    float*    fB    = (float*)   (smem + OFF_BIAS);
    float* fbE1 = fB;        float* fbE2 = fB + 64;  float* fbM = fB + 128;
    float* fbU  = fB + 256;  float* fwA  = fB + 512; float* fbA = fB + 576;

    const int tid  = threadIdx.x;
    const int lane = tid & 31;
    // SGPR wave id: guarantees uniform (scalar) branches so EXEC stays all-1s
    // around every WMMA (CDNA5 requires EXEC==all-ones; WMMA is not skipped
    // when EXEC==0).
    const int wave = __builtin_amdgcn_readfirstlane(tid) >> 5;
    const int tiles = L_SZ / 16;
    const int b  = blockIdx.x / tiles;
    const int l0 = (blockIdx.x % tiles) * 16;
    const float* hb = hin + (size_t)b * L_SZ * D_SZ;

    // ---- cooperative stage: phase-1 weights (f32 -> f16, transposed to NxK) ----
    for (int i = tid; i < E_SZ * D_SZ; i += 256) {
        int e = i / D_SZ, d = i % D_SZ;
        sWe1s[e * D_SZ + d] = (_Float16)w_e1[d * E_SZ + e];
        sWe1n[e * D_SZ + d] = (_Float16)w_e1[(D_SZ + d) * E_SZ + e];
    }
    for (int i = tid; i < E_SZ * E_SZ; i += 256) {
        int o = i / E_SZ, e = i % E_SZ;
        sWe2[o * E_SZ + e] = (_Float16)w_e2[e * E_SZ + o];
    }
    for (int i = tid; i < D_SZ * D_SZ; i += 256) {
        int m = i / D_SZ, d = i % D_SZ;
        sWmn[m * D_SZ + d] = (_Float16)w_m[d * D_SZ + m];
    }
    for (int i = tid; i < D_SZ * E_SZ; i += 256) {
        int m = i / E_SZ, e = i % E_SZ;
        sWme[m * E_SZ + e] = (_Float16)w_m[(D_SZ + e) * D_SZ + m];
    }
    if (tid < 64)  { fbE1[tid] = b_e1[tid]; fbE2[tid] = b_e2[tid]; fwA[tid] = w_a[tid]; }
    if (tid < 128) fbM[tid] = b_m[tid];
    if (tid < 256) fbU[tid] = b_u[tid];
    if (tid == 0)  fbA[0] = b_a[0];
    // self rows -> sHS (f16) and sCA[:,0:128]
    for (int i = tid; i < 16 * D_SZ; i += 256) {
        int r = i / D_SZ, d = i % D_SZ;
        _Float16 v = (_Float16)hb[(size_t)(l0 + r) * D_SZ + d];
        sHS[i] = v;
        sCA[r * 256 + d] = v;
    }
    // gather neighbor rows (float4, coalesced) -> sHN f16
    for (int i = tid; i < 256 * (D_SZ / 4); i += 256) {
        int row = i / (D_SZ / 4), c = i % (D_SZ / 4);
        int node = l0 + (row >> 4), j = row & 15;
        int idx = nb_index(node, j);
        const float4 v = *(const float4*)(hb + (size_t)idx * D_SZ + c * 4);
        _Float16* dst = sHN + row * D_SZ + c * 4;
        dst[0] = (_Float16)v.x; dst[1] = (_Float16)v.y;
        dst[2] = (_Float16)v.z; dst[3] = (_Float16)v.w;
    }
    __syncthreads();

    // ---- phase 1a: eSelf (16x64) = hS @ w_e1_self : waves 0..3, one N-tile each
    if (wave < 4) {   // uniform scalar branch (wave is SGPR)
        const int nt = wave;
        v8f acc = {};
#pragma unroll
        for (int kc = 0; kc < 4; ++kc) {
            v16h a  = ld_frag(sHS, D_SZ, kc, lane);
            v16h bf = ld_frag(sWe1s + nt * 16 * D_SZ, D_SZ, kc, lane);
            acc = wmma_step(a, bf, acc);
        }
        const int hh = lane >> 4, col = lane & 15;
#pragma unroll
        for (int j = 0; j < 8; ++j)
            sES[(j + 8 * hh) * E_SZ + nt * 16 + col] = (_Float16)acc[j];
    }
    __syncthreads();

    // ---- phase 1b: each wave owns 2 nodes (2 M-tiles of the 256-row K-batch)
    for (int s = 0; s < 2; ++s) {
        const int m  = wave * 2 + s;
        const int mb = m * 16;
        const int hh = lane >> 4, col = lane & 15;

        // e_pre = hN@w_e1_nb + eSelf + b_e1 ; gelu -> sEG
        for (int nt = 0; nt < 4; ++nt) {
            v8f acc = {};
#pragma unroll
            for (int kc = 0; kc < 4; ++kc) {
                v16h a  = ld_frag(sHN + mb * D_SZ, D_SZ, kc, lane);
                v16h bf = ld_frag(sWe1n + nt * 16 * D_SZ, D_SZ, kc, lane);
                acc = wmma_step(a, bf, acc);
            }
            const int e = nt * 16 + col;
            const float add = (float)sES[m * E_SZ + e] + fbE1[e];
#pragma unroll
            for (int j = 0; j < 8; ++j)
                sEG[(mb + j + 8 * hh) * E_SZ + e] = (_Float16)gelu_exact(acc[j] + add);
        }
        // edge = gelu(e_pre)@w_e2 + b_e2 -> sED
        for (int nt = 0; nt < 4; ++nt) {
            v8f acc = {};
#pragma unroll
            for (int kc = 0; kc < 2; ++kc) {
                v16h a  = ld_frag(sEG + mb * E_SZ, E_SZ, kc, lane);
                v16h bf = ld_frag(sWe2 + nt * 16 * E_SZ, E_SZ, kc, lane);
                acc = wmma_step(a, bf, acc);
            }
            const int e = nt * 16 + col;
#pragma unroll
            for (int j = 0; j < 8; ++j)
                sED[(mb + j + 8 * hh) * E_SZ + e] = (_Float16)(acc[j] + fbE2[e]);
        }
        // attention: logit per neighbor, softmax over the 16 lanes (both halves dup)
        {
            const int t = lane & 15;
            float logit = fbA[0];
            const _Float16* er = sED + (mb + t) * E_SZ;
            for (int e = 0; e < E_SZ; ++e) logit += (float)er[e] * fwA[e];
            float mx = logit;
#pragma unroll
            for (int d = 1; d < 16; d <<= 1) mx = fmaxf(mx, __shfl_xor(mx, d, 32));
            float p = expf(logit - mx);
            float sm = p;
#pragma unroll
            for (int d = 1; d < 16; d <<= 1) sm += __shfl_xor(sm, d, 32);
            if (lane < 16) sAT[m * 16 + t] = p / sm;
        }
        // messages = gelu(hN@w_m_nb + edge@w_m_e + b_m); agg = attn-weighted sum -> sCA[:,128:]
        for (int nt = 0; nt < 8; ++nt) {
            v8f acc = {};
#pragma unroll
            for (int kc = 0; kc < 4; ++kc) {
                v16h a  = ld_frag(sHN + mb * D_SZ, D_SZ, kc, lane);
                v16h bf = ld_frag(sWmn + nt * 16 * D_SZ, D_SZ, kc, lane);
                acc = wmma_step(a, bf, acc);
            }
#pragma unroll
            for (int kc = 0; kc < 2; ++kc) {
                v16h a  = ld_frag(sED + mb * E_SZ, E_SZ, kc, lane);
                v16h bf = ld_frag(sWme + nt * 16 * E_SZ, E_SZ, kc, lane);
                acc = wmma_step(a, bf, acc);
            }
            const int d = nt * 16 + col;
            const float bm = fbM[d];
            float part = 0.f;
#pragma unroll
            for (int j = 0; j < 8; ++j)
                part += gelu_exact(acc[j] + bm) * sAT[m * 16 + 8 * hh + j];
            part += __shfl_xor(part, 16, 32);
            if (lane < 16) sCA[m * 256 + 128 + d] = (_Float16)part;
        }
    }
    __syncthreads();

    // ---- phase 2: overlay w_u^T (256 out x 256 in) over phase-1 weight block
    for (int i = tid; i < 256 * 256; i += 256) {
        int o = i >> 8, ii = i & 255;
        sWu[o * 256 + ii] = (_Float16)w_u[ii * 256 + o];
    }
    __syncthreads();

    // upd = [h|agg] @ w_u + b_u : each wave does 2 of 16 N-tiles
    float* sUPD = (float*)sED;   // 16x256 f32 reuse
    for (int s = 0; s < 2; ++s) {
        const int nt = wave * 2 + s;
        v8f acc = {};
#pragma unroll
        for (int kc = 0; kc < 8; ++kc) {
            v16h a  = ld_frag(sCA, 256, kc, lane);
            v16h bf = ld_frag(sWu + nt * 16 * 256, 256, kc, lane);
            acc = wmma_step(a, bf, acc);
        }
        const int hh = lane >> 4, col = lane & 15;
        const int o = nt * 16 + col;
        const float bu = fbU[o];
#pragma unroll
        for (int j = 0; j < 8; ++j)
            sUPD[(j + 8 * hh) * 256 + o] = acc[j] + bu;
    }
    __syncthreads();

    // gate/LN/residual: wave w -> rows 2w, 2w+1 ; lane handles 4 columns
    for (int s = 0; s < 2; ++s) {
        const int r = wave * 2 + s;
        float v0[4];
        float mean = 0.f;
#pragma unroll
        for (int q = 0; q < 4; ++q) { v0[q] = sUPD[r * 256 + 128 + lane * 4 + q]; mean += v0[q]; }
#pragma unroll
        for (int d = 1; d < 32; d <<= 1) mean += __shfl_xor(mean, d, 32);
        mean *= (1.0f / 128.0f);
        float var = 0.f;
#pragma unroll
        for (int q = 0; q < 4; ++q) { float t = v0[q] - mean; var += t * t; }
#pragma unroll
        for (int d = 1; d < 32; d <<= 1) var += __shfl_xor(var, d, 32);
        var *= (1.0f / 128.0f);
        const float rstd = rsqrtf(var + 1e-5f);
#pragma unroll
        for (int q = 0; q < 4; ++q) {
            const int d = lane * 4 + q;
            const float ln   = (v0[q] - mean) * rstd * g_un[d] + b_un[d];
            const float gate = 1.0f / (1.0f + expf(-sUPD[r * 256 + d]));
            const float hold = hb[(size_t)(l0 + r) * D_SZ + d];
            hout[((size_t)b * L_SZ + l0 + r) * D_SZ + d] = gate * ln + (1.0f - gate) * hold;
        }
    }
}

__global__ __launch_bounds__(256)
void graph_final_kernel(const float* __restrict__ h,
                        const float* __restrict__ g_n, const float* __restrict__ b_n,
                        const float* __restrict__ w_o, float* __restrict__ out) {
    __shared__ _Float16 sWo[D_SZ * D_SZ];
    __shared__ _Float16 sA[16 * D_SZ];
    const int tid = threadIdx.x, lane = tid & 31;
    const int wave = __builtin_amdgcn_readfirstlane(tid) >> 5;  // SGPR wave id
    const int tiles = L_SZ / 16;
    const int b  = blockIdx.x / tiles;
    const int l0 = (blockIdx.x % tiles) * 16;
    const float* hb = h + ((size_t)b * L_SZ + l0) * D_SZ;

    for (int i = tid; i < D_SZ * D_SZ; i += 256) {
        int o = i >> 7, d = i & 127;
        sWo[o * D_SZ + d] = (_Float16)w_o[d * D_SZ + o];
    }
    for (int s = 0; s < 2; ++s) {
        const int r = wave * 2 + s;
        float v0[4];
        float mean = 0.f;
#pragma unroll
        for (int q = 0; q < 4; ++q) { v0[q] = hb[r * D_SZ + lane * 4 + q]; mean += v0[q]; }
#pragma unroll
        for (int d = 1; d < 32; d <<= 1) mean += __shfl_xor(mean, d, 32);
        mean *= (1.0f / 128.0f);
        float var = 0.f;
#pragma unroll
        for (int q = 0; q < 4; ++q) { float t = v0[q] - mean; var += t * t; }
#pragma unroll
        for (int d = 1; d < 32; d <<= 1) var += __shfl_xor(var, d, 32);
        var *= (1.0f / 128.0f);
        const float rstd = rsqrtf(var + 1e-5f);
#pragma unroll
        for (int q = 0; q < 4; ++q) {
            const int d = lane * 4 + q;
            sA[r * D_SZ + d] = (_Float16)((v0[q] - mean) * rstd * g_n[d] + b_n[d]);
        }
    }
    __syncthreads();

    const int nt = wave;   // 8 N-tiles of 16 output columns
    v8f acc = {};
#pragma unroll
    for (int kc = 0; kc < 4; ++kc) {
        v16h a  = ld_frag(sA, D_SZ, kc, lane);
        v16h bf = ld_frag(sWo + nt * 16 * D_SZ, D_SZ, kc, lane);
        acc = wmma_step(a, bf, acc);
    }
    const int hh = lane >> 4, col = lane & 15, o = nt * 16 + col;
#pragma unroll
    for (int j = 0; j < 8; ++j)
        out[((size_t)b * L_SZ + l0 + j + 8 * hh) * D_SZ + o] = acc[j];
}

extern "C" void kernel_launch(void* const* d_in, const int* in_sizes, int n_in,
                              void* d_out, int out_size, void* d_ws, size_t ws_size,
                              hipStream_t stream) {
    const float* x    = (const float*)d_in[0];
    const float* w_e1 = (const float*)d_in[1];
    const float* b_e1 = (const float*)d_in[2];
    const float* w_e2 = (const float*)d_in[3];
    const float* b_e2 = (const float*)d_in[4];
    const float* w_m  = (const float*)d_in[5];
    const float* b_m  = (const float*)d_in[6];
    const float* w_a  = (const float*)d_in[7];
    const float* b_a  = (const float*)d_in[8];
    const float* w_u  = (const float*)d_in[9];
    const float* b_u  = (const float*)d_in[10];
    const float* g_un = (const float*)d_in[11];
    const float* b_un = (const float*)d_in[12];
    const float* g_n  = (const float*)d_in[13];
    const float* b_n  = (const float*)d_in[14];
    const float* w_o  = (const float*)d_in[15];

    float* h0 = (float*)d_ws;
    float* h1 = h0 + (size_t)B_SZ * L_SZ * D_SZ;

    (void)in_sizes; (void)n_in; (void)out_size; (void)ws_size;

    hipFuncSetAttribute((const void*)graph_iter_kernel,
                        hipFuncAttributeMaxDynamicSharedMemorySize, SMEM_TOTAL);

    dim3 grid(B_SZ * (L_SZ / 16)), block(256);
    graph_iter_kernel<<<grid, block, SMEM_TOTAL, stream>>>(
        x, h0, w_e1, b_e1, w_e2, b_e2, w_m, b_m, w_a, b_a, w_u, b_u, g_un, b_un);
    graph_iter_kernel<<<grid, block, SMEM_TOTAL, stream>>>(
        h0, h1, w_e1, b_e1, w_e2, b_e2, w_m, b_m, w_a, b_a, w_u, b_u, g_un, b_un);
    graph_final_kernel<<<grid, block, 0, stream>>>(h1, g_n, b_n, w_o, (float*)d_out);
}
